// DynamicConv1d_21191368639260
// MI455X (gfx1250) — compile-verified, matrix operand order checked by reading, not run on hardware
//
#include <hip/hip_runtime.h>
#include <hip/hip_fp16.h>

#define BS     64
#define C_IN   64
#define LLEN   4096
#define C_OUT  128
#define KS     7
#define KBANK  4
#define HIDDEN 16
#define PADW   3
#define TEMP   30.0f
#define KDIM   (C_IN * KS)          // 448

#define BM 32                        // out-channel tile
#define BN 128                       // L tile
#define XSTRIDE 72                   // halves per l-row in LDS (16B-aligned runs)
#define XROWS (BN + KS - 1)          // 134

#define AGGW_HALFS ((size_t)BS * C_OUT * KDIM)     // 3,670,016 halves
#define AGGB_OFF_BYTES (AGGW_HALFS * 2)            // 7,340,032 bytes

typedef _Float16 half16 __attribute__((ext_vector_type(16)));
typedef _Float16 half8  __attribute__((ext_vector_type(8)));
typedef float    f32x8  __attribute__((ext_vector_type(8)));
typedef unsigned int uint32x4 __attribute__((ext_vector_type(4)));
typedef int          int32x8  __attribute__((ext_vector_type(8)));
typedef int          int32x4  __attribute__((ext_vector_type(4)));

#if __has_builtin(__builtin_amdgcn_tensor_load_to_lds) && \
    __has_builtin(__builtin_amdgcn_s_wait_tensorcnt)
#define USE_TDM 1
#else
#define USE_TDM 0
#endif

// ---------------------------------------------------------------------------
// Phase 1: attention + weight/bias aggregation (one block per batch sample).
// Writes aggW (f16, [b][o][f][i]) and aggB (f32, [b][o]) into workspace.
// ---------------------------------------------------------------------------
__global__ __launch_bounds__(256) void attn_agg_kernel(
    const float* __restrict__ x,
    const float* __restrict__ w1,     // [HIDDEN, C_IN]
    const float* __restrict__ w2,     // [KBANK, HIDDEN]
    const float* __restrict__ wgt,    // [KBANK, C_OUT, C_IN, KS]
    const float* __restrict__ bias,   // [KBANK, C_OUT]
    _Float16* __restrict__ aggW,
    float* __restrict__ aggB) {
  const int b = blockIdx.x;
  const int tid = threadIdx.x;

  __shared__ float sPool[C_IN];
  __shared__ float sH[HIDDEN];
  __shared__ float sAtt[KBANK];

  if (tid < C_IN) sPool[tid] = 0.0f;
  __syncthreads();

  // ---- global average pool: coalesced float4 reads, LDS atomic reduce ----
  {
    const float4* xb = (const float4*)(x + (size_t)b * C_IN * LLEN);
    const int NV4 = C_IN * (LLEN / 4);           // 65536 vec4
    for (int base = tid * 4; base < NV4; base += 256 * 4) {
      float s = 0.0f;
#pragma unroll
      for (int j = 0; j < 4; ++j) {
        float4 v = xb[base + j];
        s += v.x + v.y + v.z + v.w;
      }
      atomicAdd(&sPool[base / (LLEN / 4)], s);   // 4 consecutive vec4 share channel
    }
  }
  __syncthreads();

  // ---- h = relu(pooled @ w1^T) ----
  if (tid < HIDDEN) {
    float acc = 0.0f;
#pragma unroll
    for (int c = 0; c < C_IN; ++c)
      acc += (sPool[c] * (1.0f / LLEN)) * w1[tid * C_IN + c];
    sH[tid] = acc > 0.0f ? acc : 0.0f;
  }
  __syncthreads();

  // ---- logits, softmax(./T) ----
  if (tid == 0) {
    float lg[KBANK];
    float mx = -3.0e38f;
#pragma unroll
    for (int k = 0; k < KBANK; ++k) {
      float a = 0.0f;
#pragma unroll
      for (int j = 0; j < HIDDEN; ++j) a += sH[j] * w2[k * HIDDEN + j];
      lg[k] = a * (1.0f / TEMP);
      mx = fmaxf(mx, lg[k]);
    }
    float den = 0.0f;
#pragma unroll
    for (int k = 0; k < KBANK; ++k) { lg[k] = __expf(lg[k] - mx); den += lg[k]; }
    float inv = 1.0f / den;
#pragma unroll
    for (int k = 0; k < KBANK; ++k) sAtt[k] = lg[k] * inv;
  }
  __syncthreads();

  const float a0 = sAtt[0], a1 = sAtt[1], a2 = sAtt[2], a3 = sAtt[3];

  // ---- aggregated bias ----
  if (tid < C_OUT) {
    aggB[b * C_OUT + tid] = a0 * bias[0 * C_OUT + tid] + a1 * bias[1 * C_OUT + tid]
                          + a2 * bias[2 * C_OUT + tid] + a3 * bias[3 * C_OUT + tid];
  }

  // ---- aggregated weights -> f16, layout [o][f][i] so K=f*64+i is contiguous ----
  _Float16* wOut = aggW + (size_t)b * C_OUT * KDIM;
  const size_t bankStride = (size_t)C_OUT * C_IN * KS;   // 57344
  for (int e = tid; e < C_OUT * KDIM; e += 256) {
    int o = e / KDIM;
    int r = e - o * KDIM;
    int f = r >> 6;
    int i = r & 63;
    const float* wb = wgt + (((size_t)o) * C_IN + i) * KS + f;
    float v = a0 * wb[0] + a1 * wb[bankStride] + a2 * wb[2 * bankStride] + a3 * wb[3 * bankStride];
    wOut[e] = (_Float16)v;
  }
}

// ---------------------------------------------------------------------------
// Phase 2: per-sample conv as f16 WMMA GEMM.
// Block = (32 out-ch) x (128 L), 8 waves; wave w owns N-strip w*16.
// W tile staged via the Tensor Data Mover (single TDM descriptor, 28,672 B),
// overlapped with the x-tile f32->f16 transpose into LDS.
// ---------------------------------------------------------------------------
__global__ __launch_bounds__(256) void dynconv_wmma_kernel(
    const float* __restrict__ x,
    const _Float16* __restrict__ aggW,
    const float* __restrict__ aggB,
    float* __restrict__ out) {
  __shared__ __align__(16) _Float16 sW[BM * KDIM];        // 28,672 B
  __shared__ __align__(16) _Float16 sX[XROWS * XSTRIDE];  // 19,296 B

  const int tid   = threadIdx.x;
  const int lTile = blockIdx.x;     // 0..31
  const int oTile = blockIdx.y;     // 0..3
  const int b     = blockIdx.z;     // 0..63
  const int L0    = lTile * BN;
  const int oBase = oTile * BM;

  // ---- stage W tile [32 x 448] halves ----
#if USE_TDM
  if (tid < 32) {   // one wave issues the TDM DMA (EXEC ignored by tensor ops)
    const unsigned long long gaddr =
        (unsigned long long)(const void*)(aggW + ((size_t)b * C_OUT + oBase) * KDIM);
    const unsigned ldsaddr = (unsigned)(unsigned long long)(const void*)&sW[0];
    const unsigned NELEM = (BM * KDIM * 2) / 8;            // 3584 x 8-byte elems

    uint32x4 g0;
    g0[0] = 1u;                                            // count=1 valid D#
    g0[1] = ldsaddr;                                       // LDS dest (bytes)
    g0[2] = (unsigned)(gaddr & 0xffffffffu);               // global_addr[31:0]
    g0[3] = (unsigned)((gaddr >> 32) & 0x01ffffffu)        // global_addr[56:32]
          | (2u << 30);                                    // type = 2 (image)

    int32x8 g1;
    g1[0] = (3 << 16);                  // workgroup_mask=0, data_size=3 (8B)
    g1[1] = (int)((NELEM & 0xffffu) << 16);  // tensor_dim0[15:0] @ bits 63:48
    g1[2] = (int)(1u << 16);            // tensor_dim0 hi=0, tensor_dim1=1
    g1[3] = (int)((NELEM & 0xffffu) << 16);  // tile_dim0 = 3584 @ bits 127:112
    g1[4] = 0;                          // tile_dim1=0, tile_dim2=0 (1-D tile)
    g1[5] = (int)NELEM;                 // tensor_dim0_stride (low 32)
    g1[6] = 0;
    g1[7] = 0;

    int32x4 z4 = {0, 0, 0, 0};          // groups 2/3 unused (<=2-D tensor)
    int32x8 z8 = {0, 0, 0, 0, 0, 0, 0, 0};
    __builtin_amdgcn_tensor_load_to_lds(g0, g1, z4, z4, z8, 0);
  }
#else
  {
    const uint4* src = (const uint4*)(aggW + ((size_t)b * C_OUT + oBase) * KDIM);
    uint4* dst = (uint4*)sW;
    for (int idx = tid; idx < (BM * KDIM) / 8; idx += 256) dst[idx] = src[idx];
  }
#endif

  // ---- stage X tile with halo, transposed: sX[lloc][i] = x[b][i][L0+lloc-3] ----
  for (int e = tid; e < C_IN * XROWS; e += 256) {
    int i = e / XROWS;
    int lloc = e - i * XROWS;
    int lg = L0 + lloc - PADW;
    float v = (lg >= 0 && lg < LLEN) ? x[((size_t)b * C_IN + i) * LLEN + lg] : 0.0f;
    sX[lloc * XSTRIDE + i] = (_Float16)v;
  }

#if USE_TDM
  __builtin_amdgcn_s_wait_tensorcnt(0);   // W tile DMA complete (uniform, cheap)
#endif
  __syncthreads();

  const int lane = tid & 31;
  const int wv   = tid >> 5;        // wave id 0..7
  const int m    = lane & 15;       // row (A) / column (B) index
  const int hi   = lane >> 4;       // K-half select

  f32x8 acc0 = {};
  f32x8 acc1 = {};

#pragma unroll
  for (int kk = 0; kk < KDIM / 32; ++kk) {     // 14 K-chunks of 32
    const int f  = kk >> 1;
    const int i0 = (kk & 1) << 5;
    const int kb = kk << 5;

    // A fragments: sW[row][K], per lane two contiguous 8-half runs (ds_load_b128)
    const _Float16* pa0 = sW + (0 * 16 + m) * KDIM + kb + 8 * hi;
    const _Float16* pa1 = sW + (1 * 16 + m) * KDIM + kb + 8 * hi;
    half8 a0lo = *(const half8*)pa0;
    half8 a0hi = *(const half8*)(pa0 + 16);
    half8 a1lo = *(const half8*)pa1;
    half8 a1hi = *(const half8*)(pa1 + 16);

    // B fragment: column n = m -> l fixed per lane, K runs contiguous in channel i
    const int lloc = wv * 16 + m + f;
    const _Float16* pb = sX + lloc * XSTRIDE + i0 + 8 * hi;
    half8 blo = *(const half8*)pb;
    half8 bhi = *(const half8*)(pb + 16);

    half16 A0, A1, B;
#pragma unroll
    for (int e = 0; e < 8; ++e) {
      A0[e] = a0lo[e]; A0[e + 8] = a0hi[e];
      A1[e] = a1lo[e]; A1[e + 8] = a1hi[e];
      B[e]  = blo[e];  B[e + 8]  = bhi[e];
    }
    acc0 = __builtin_amdgcn_wmma_f32_16x16x32_f16(false, A0, false, B, (short)0, acc0, false, false);
    acc1 = __builtin_amdgcn_wmma_f32_16x16x32_f16(false, A1, false, B, (short)0, acc1, false, false);
  }

  // ---- epilogue: + bias, coalesced f32 stores ----
  const int n  = lane & 15;
  const int lg = L0 + wv * 16 + n;
#pragma unroll
  for (int r = 0; r < 8; ++r) {
    int o0 = oBase + 0 * 16 + r + 8 * hi;   // C/D layout: VGPR r -> M = r + 8*hi
    int o1 = oBase + 1 * 16 + r + 8 * hi;
    out[((size_t)b * C_OUT + o0) * LLEN + lg] = acc0[r] + aggB[b * C_OUT + o0];
    out[((size_t)b * C_OUT + o1) * LLEN + lg] = acc1[r] + aggB[b * C_OUT + o1];
  }
}

// ---------------------------------------------------------------------------
extern "C" void kernel_launch(void* const* d_in, const int* in_sizes, int n_in,
                              void* d_out, int out_size, void* d_ws, size_t ws_size,
                              hipStream_t stream) {
  const float* x    = (const float*)d_in[0];
  const float* w1   = (const float*)d_in[1];
  const float* w2   = (const float*)d_in[2];
  const float* wgt  = (const float*)d_in[3];
  const float* bias = (const float*)d_in[4];

  _Float16* aggW = (_Float16*)d_ws;
  float*    aggB = (float*)((char*)d_ws + AGGB_OFF_BYTES);
  float*    out  = (float*)d_out;

  hipLaunchKernelGGL(attn_agg_kernel, dim3(BS), dim3(256), 0, stream,
                     x, w1, w2, wgt, bias, aggW, aggB);
  hipLaunchKernelGGL(dynconv_wmma_kernel, dim3(LLEN / BN, C_OUT / BM, BS),
                     dim3(256), 0, stream, x, aggW, aggB, out);
}
